// CSSMSHViT_60043642798201
// MI455X (gfx1250) — matrix operand order repaired and checked by reading, not compiled, not run on hardware
//
#include <hip/hip_runtime.h>
#include <math.h>

// ---------------- problem constants ----------------
#define B_   32
#define T_   8
#define H_   16
#define W_   16
#define C_   384
#define HID_ 1536
#define GH_  96
#define P_   (H_ * W_)             // 256 spatial positions
#define NROWS_ (B_ * P_)           // 8192 rows for big GEMMs
#define SZ_  (NROWS_ * C_)         // 3,145,728
#define HSZ_ ((size_t)NROWS_ * HID_) // 12,582,912
#define RHO_ 0.999f
#define EPS_ 1e-6f

typedef __attribute__((ext_vector_type(16))) _Float16 v16h;
typedef __attribute__((ext_vector_type(8)))  _Float16 v8h;
typedef __attribute__((ext_vector_type(8)))  float    v8f;

__device__ __forceinline__ float sigmoid_f(float x) { return 1.0f / (1.0f + expf(-x)); }
__device__ __forceinline__ float gelu_f(float x) {
    const float k0 = 0.7978845608028654f; // sqrt(2/pi)
    return 0.5f * x * (1.0f + tanhf(k0 * (x + 0.044715f * x * x * x)));
}

// ---------------- WMMA GEMM: out[N,Co] = act(in[N,K] @ W[K,Co] + bias) (+resid) ----------------
// Block tile 128x64, K-step 32. 8 waves; wave wv owns rows [wv*16, wv*16+16) x all 64 cols
// (4 accumulators -> 4 v_wmma per K-step). Register-prefetch software pipeline: next K-slab's
// global loads are issued right after the barrier and retire while WMMAs run on the current slab.
// EPI: 0=none, 1=rho*sigmoid, 2=silu, 3=+resid
template <int EPI>
__global__ __launch_bounds__(256)
void gemm_wmma_f16(const float* __restrict__ A, const float* __restrict__ Wm,
                   const float* __restrict__ bias, const float* __restrict__ resid,
                   float* __restrict__ out, int Nrows, int K, int Ncols)
{
    __shared__ alignas(16) _Float16 As[128][40];  // [row][k], padded stride
    __shared__ alignas(16) _Float16 Bs[64][40];   // [col][k] (transposed)

    const int tid  = threadIdx.x;
    const int lane = tid & 31;
    const int wv   = tid >> 5;            // 0..7 row strip
    const int row0 = blockIdx.y * 128;
    const int col0 = blockIdx.x * 64;

    v8f acc[4] = {};

    // staging assignment: A 128x32 (16 elems/thread), B 32x64 (8 elems/thread)
    const int la_row = tid >> 1;          // 0..127
    const int la_kc  = (tid & 1) * 16;    // 0,16
    const int lb_n   = tid & 63;          // 0..63
    const int lb_k0  = (tid >> 6) * 8;    // 0,8,16,24

    const float* aptr = A + (size_t)(row0 + la_row) * K + la_kc;

    float areg[16];
    float breg[8];
    // prefetch K-slab 0 into registers
    #pragma unroll
    for (int j = 0; j < 16; ++j) areg[j] = aptr[j];
    #pragma unroll
    for (int j = 0; j < 8; ++j)
        breg[j] = Wm[(size_t)(lb_k0 + j) * Ncols + (col0 + lb_n)];

    for (int k0 = 0; k0 < K; k0 += 32) {
        // drain prefetch regs -> LDS (f32 -> f16)
        {
            _Float16* dst = &As[la_row][la_kc];
            #pragma unroll
            for (int j = 0; j < 16; ++j) dst[j] = (_Float16)areg[j];
            #pragma unroll
            for (int j = 0; j < 8; ++j) Bs[lb_n][lb_k0 + j] = (_Float16)breg[j];
        }
        __syncthreads();

        // issue next K-slab's global loads; they complete while WMMAs execute
        if (k0 + 32 < K) {
            const float* ap2 = aptr + (k0 + 32);
            #pragma unroll
            for (int j = 0; j < 16; ++j) areg[j] = ap2[j];
            #pragma unroll
            for (int j = 0; j < 8; ++j)
                breg[j] = Wm[(size_t)(k0 + 32 + lb_k0 + j) * Ncols + (col0 + lb_n)];
        }

        // A fragment: lane<16 -> kb=0, lane>=16 -> kb=8; e<8: k=kb+e, e>=8: k=kb+e+8
        const int m  = wv * 16 + (lane & 15);
        const int kb = (lane >= 16) ? 8 : 0;
        v8h a_lo = *(const v8h*)&As[m][kb];
        v8h a_hi = *(const v8h*)&As[m][kb + 16];
        v16h afrag;
        #pragma unroll
        for (int e = 0; e < 8; ++e) { afrag[e] = a_lo[e]; afrag[e + 8] = a_hi[e]; }

        // B fragments: lane holds column n, kb=(lane>=16?8:0); e<8: k=kb+e, e>=8: k=kb+e+8
        const int nb  = lane & 15;
        const int kk0 = (lane >= 16) ? 8 : 0;
        #pragma unroll
        for (int tc = 0; tc < 4; ++tc) {
            v8h blo = *(const v8h*)&Bs[tc * 16 + nb][kk0];
            v8h bhi = *(const v8h*)&Bs[tc * 16 + nb][kk0 + 16];
            v16h bfrag;
            #pragma unroll
            for (int e = 0; e < 8; ++e) { bfrag[e] = blo[e]; bfrag[e + 8] = bhi[e]; }
            acc[tc] = __builtin_amdgcn_wmma_f32_16x16x32_f16(false, afrag, false, bfrag,
                                                             (short)0, acc[tc], false, false);
        }
        __syncthreads();
    }

    // epilogue: D layout row = r + (lane>=16?8:0), col = lane&15 (coalesced stores)
    const int rbase = row0 + wv * 16 + ((lane >= 16) ? 8 : 0);
    #pragma unroll
    for (int tc = 0; tc < 4; ++tc) {
        const int c = col0 + tc * 16 + (lane & 15);
        const float bz = bias[c];
        #pragma unroll
        for (int r = 0; r < 8; ++r) {
            const size_t idx = (size_t)(rbase + r) * Ncols + c;
            float v = acc[tc][r] + bz;
            if (EPI == 1) v = RHO_ * sigmoid_f(v);
            else if (EPI == 2) v = v * sigmoid_f(v);
            else if (EPI == 3) v += resid[idx];
            out[idx] = v;
        }
    }
}

// ---------------- per-batch global-LN stats + per-channel spatial mean q ----------------
__global__ __launch_bounds__(384)
void stats_kernel(const float* __restrict__ x, float* __restrict__ qmean,
                  float* __restrict__ meanb, float* __restrict__ rstdb)
{
    int b = blockIdx.x, c = threadIdx.x;
    const float* xb = x + (size_t)b * P_ * C_;
    float s1 = 0.f, s2 = 0.f;
    for (int p = 0; p < P_; ++p) { float v = xb[p * C_ + c]; s1 += v; s2 += v * v; }
    qmean[b * C_ + c] = s1 * (1.0f / P_);
    __shared__ float sh1[384], sh2[384];
    sh1[c] = s1; sh2[c] = s2;
    __syncthreads();
    if (c < 128) { sh1[c] += sh1[c + 128] + sh1[c + 256]; sh2[c] += sh2[c + 128] + sh2[c + 256]; }
    __syncthreads();
    for (int s = 64; s > 0; s >>= 1) {
        if (c < s) { sh1[c] += sh1[c + s]; sh2[c] += sh2[c + s]; }
        __syncthreads();
    }
    if (c == 0) {
        float n = (float)(P_ * C_);
        float m = sh1[0] / n;
        float var = sh2[0] / n - m * m;
        meanb[b] = m;
        rstdb[b] = rsqrtf(var + EPS_);
    }
}

__global__ void ln1_kernel(const float* __restrict__ x, const float* __restrict__ meanb,
                           const float* __restrict__ rstdb, const float* __restrict__ gamma,
                           const float* __restrict__ beta, float* __restrict__ out)
{
    int i = blockIdx.x * blockDim.x + threadIdx.x;
    if (i >= SZ_) return;
    int c = i % C_;
    int b = i / (P_ * C_);
    out[i] = (x[i] - meanb[b]) * rstdb[b] * gamma[c] + beta[c];
}

// xn = xnln + dwconv3x3(xnln) + b
__global__ void posconv_kernel(const float* __restrict__ xin, const float* __restrict__ w,
                               const float* __restrict__ bc, float* __restrict__ out)
{
    int i = blockIdx.x * blockDim.x + threadIdx.x;
    if (i >= SZ_) return;
    int c = i % C_; int pos = i / C_;
    int xx = pos % W_, yy = (pos / W_) % H_, b = pos / P_;
    const float* base = xin + (size_t)b * P_ * C_;
    float acc = 0.f;
    #pragma unroll
    for (int ky = 0; ky < 3; ++ky)
        #pragma unroll
        for (int kx = 0; kx < 3; ++kx) {
            int y = yy + ky - 1, x2 = xx + kx - 1;
            if (y >= 0 && y < H_ && x2 >= 0 && x2 < W_)
                acc += w[(ky * 3 + kx) * C_ + c] * base[(y * W_ + x2) * C_ + c];
        }
    out[i] = xin[i] + acc + bc[c];
}

// per-(b,c) block: hm_t = conv5x5((1-a^t)z)+b_sp for t=1..8, reduce mean_p(hm_t * g)
__global__ __launch_bounds__(256)
void tmeans_kernel(const float* __restrict__ a, const float* __restrict__ z,
                   const float* __restrict__ g, const float* __restrict__ ksp,
                   const float* __restrict__ bsp, float* __restrict__ kmean)
{
    int bc = blockIdx.x;
    int b = bc / C_, c = bc % C_;
    int p = threadIdx.x;
    int xx = p % W_, yy = p / W_;
    const size_t bo = (size_t)b * P_ * C_;
    float acc[T_];
    #pragma unroll
    for (int t = 0; t < T_; ++t) acc[t] = 0.f;
    for (int ky = 0; ky < 5; ++ky)
        for (int kx = 0; kx < 5; ++kx) {
            int y = yy + ky - 2, x2 = xx + kx - 2;
            if (y >= 0 && y < H_ && x2 >= 0 && x2 < W_) {
                size_t idx = bo + (size_t)(y * W_ + x2) * C_ + c;
                float aq = a[idx], zq = z[idx];
                float wgt = ksp[(ky * 5 + kx) * C_ + c];
                float ap = 1.f;
                #pragma unroll
                for (int t = 0; t < T_; ++t) { ap *= aq; acc[t] += wgt * (1.f - ap) * zq; }
            }
        }
    float gv = g[bo + (size_t)p * C_ + c];
    float bspc = bsp[c];
    __shared__ float sh[256];
    for (int t = 0; t < T_; ++t) {
        sh[p] = (acc[t] + bspc) * gv;
        __syncthreads();
        for (int s = 128; s > 0; s >>= 1) {
            if (p < s) sh[p] += sh[p + s];
            __syncthreads();
        }
        if (p == 0) kmean[(size_t)(b * T_ + t) * C_ + c] = sh[0] * (1.0f / P_);
        __syncthreads();
    }
}

// gate MLP: logits[b,t] from concat(q[b], k[b,t]) @ Wg1 -> gelu -> @ Wg2
__global__ __launch_bounds__(GH_)
void gate_kernel(const float* __restrict__ qmean, const float* __restrict__ ksm,
                 const float* __restrict__ Wg1, const float* __restrict__ bg1,
                 const float* __restrict__ Wg2, const float* __restrict__ bg2,
                 float* __restrict__ logits)
{
    int bt = blockIdx.x;
    int b = bt / T_;
    int j = threadIdx.x;
    const float* qb = qmean + (size_t)b * C_;
    const float* kb = ksm + (size_t)bt * C_;
    float s = bg1[j];
    for (int i = 0; i < C_; ++i) s += qb[i] * Wg1[i * GH_ + j];
    for (int i = 0; i < C_; ++i) s += kb[i] * Wg1[(C_ + i) * GH_ + j];
    __shared__ float sh[GH_];
    sh[j] = gelu_f(s) * Wg2[j];
    __syncthreads();
    if (j < 32) sh[j] += sh[j + 32] + sh[j + 64];
    __syncthreads();
    if (j < 16) sh[j] += sh[j + 16];
    __syncthreads();
    if (j < 8) sh[j] += sh[j + 8];
    __syncthreads();
    if (j < 4) sh[j] += sh[j + 4];
    __syncthreads();
    if (j < 2) sh[j] += sh[j + 2];
    __syncthreads();
    if (j == 0) logits[bt] = sh[0] + sh[1] + bg2[0];
}

__global__ __launch_bounds__(32)
void softmax_kernel(const float* __restrict__ logits, float* __restrict__ wout)
{
    int b = blockIdx.x, t = threadIdx.x;
    __shared__ float sh[T_];
    if (t < T_) sh[t] = logits[b * T_ + t] + (t == T_ - 1 ? 4.0f : 0.0f);
    __syncthreads();
    if (t == 0) {
        float mx = sh[0];
        for (int i = 1; i < T_; ++i) mx = fmaxf(mx, sh[i]);
        float sum = 0.f;
        for (int i = 0; i < T_; ++i) { sh[i] = expf(sh[i] - mx); sum += sh[i]; }
        float inv = 1.f / sum;
        for (int i = 0; i < T_; ++i) wout[b * T_ + i] = sh[i] * inv;
    }
}

// v = z * (1 - sum_t w_t a^t)
__global__ void vmix_kernel(const float* __restrict__ a, const float* __restrict__ z,
                            const float* __restrict__ wsoft, float* __restrict__ vout)
{
    int i = blockIdx.x * blockDim.x + threadIdx.x;
    if (i >= SZ_) return;
    int b = i / (P_ * C_);
    const float* wb = wsoft + b * T_;
    float aa = a[i], ap = 1.f, s = 0.f;
    #pragma unroll
    for (int t = 0; t < T_; ++t) { ap *= aa; s += wb[t] * ap; }
    vout[i] = z[i] * (1.f - s);
}

// ygate = (conv5x5(v) + b_sp) * g
__global__ void comb_kernel(const float* __restrict__ v, const float* __restrict__ g,
                            const float* __restrict__ ksp, const float* __restrict__ bsp,
                            float* __restrict__ ygate)
{
    int i = blockIdx.x * blockDim.x + threadIdx.x;
    if (i >= SZ_) return;
    int c = i % C_; int pos = i / C_;
    int xx = pos % W_, yy = (pos / W_) % H_, b = pos / P_;
    const float* base = v + (size_t)b * P_ * C_;
    float acc = 0.f;
    for (int ky = 0; ky < 5; ++ky)
        for (int kx = 0; kx < 5; ++kx) {
            int y = yy + ky - 2, x2 = xx + kx - 2;
            if (y >= 0 && y < H_ && x2 >= 0 && x2 < W_)
                acc += ksp[(ky * 5 + kx) * C_ + c] * base[(y * W_ + x2) * C_ + c];
        }
    ygate[i] = (acc + bsp[c]) * g[i];
}

// per-position LayerNorm over C
__global__ __launch_bounds__(384)
void ln2_kernel(const float* __restrict__ in, const float* __restrict__ gamma,
                const float* __restrict__ beta, float* __restrict__ out)
{
    int p = blockIdx.x, c = threadIdx.x;
    float val = in[(size_t)p * C_ + c];
    __shared__ float sh1[384], sh2[384];
    __shared__ float mv[2];
    sh1[c] = val; sh2[c] = val * val;
    __syncthreads();
    if (c < 128) { sh1[c] += sh1[c + 128] + sh1[c + 256]; sh2[c] += sh2[c + 128] + sh2[c + 256]; }
    __syncthreads();
    for (int s = 64; s > 0; s >>= 1) {
        if (c < s) { sh1[c] += sh1[c + s]; sh2[c] += sh2[c + s]; }
        __syncthreads();
    }
    if (c == 0) {
        float m = sh1[0] * (1.0f / C_);
        float var = sh2[0] * (1.0f / C_) - m * m;
        mv[0] = m; mv[1] = rsqrtf(var + EPS_);
    }
    __syncthreads();
    out[(size_t)p * C_ + c] = (val - mv[0]) * mv[1] * gamma[c] + beta[c];
}

// h1d = gelu(dwconv3x3(h1) + bdw) on HID channels
__global__ void mlpconv_kernel(const float* __restrict__ h1, const float* __restrict__ wdw,
                               const float* __restrict__ bdw, float* __restrict__ out)
{
    size_t i = (size_t)blockIdx.x * blockDim.x + threadIdx.x;
    if (i >= HSZ_) return;
    int c = (int)(i % HID_); size_t pos = i / HID_;
    int xx = (int)(pos % W_), yy = (int)((pos / W_) % H_), b = (int)(pos / P_);
    const float* base = h1 + (size_t)b * P_ * HID_;
    float acc = 0.f;
    #pragma unroll
    for (int ky = 0; ky < 3; ++ky)
        #pragma unroll
        for (int kx = 0; kx < 3; ++kx) {
            int y = yy + ky - 1, x2 = xx + kx - 1;
            if (y >= 0 && y < H_ && x2 >= 0 && x2 < W_)
                acc += wdw[(ky * 3 + kx) * HID_ + c] * base[(size_t)(y * W_ + x2) * HID_ + c];
        }
    out[i] = gelu_f(acc + bdw[c]);
}

// ---------------- orchestration ----------------
extern "C" void kernel_launch(void* const* d_in, const int* in_sizes, int n_in,
                              void* d_out, int out_size, void* d_ws, size_t ws_size,
                              hipStream_t stream)
{
    const float* x      = (const float*)d_in[0];
    const float* gamma1 = (const float*)d_in[1];
    const float* beta1  = (const float*)d_in[2];
    const float* w_pos  = (const float*)d_in[3];
    const float* b_pos  = (const float*)d_in[4];
    const float* W_in   = (const float*)d_in[5];
    const float* b_in   = (const float*)d_in[6];
    const float* W_a    = (const float*)d_in[7];
    const float* b_a    = (const float*)d_in[8];
    const float* W_g    = (const float*)d_in[9];
    const float* b_g    = (const float*)d_in[10];
    const float* k_sp   = (const float*)d_in[11];
    const float* b_sp   = (const float*)d_in[12];
    const float* W_out  = (const float*)d_in[13];
    const float* b_out  = (const float*)d_in[14];
    const float* Wg1    = (const float*)d_in[15];
    const float* bg1    = (const float*)d_in[16];
    const float* Wg2    = (const float*)d_in[17];
    const float* bg2    = (const float*)d_in[18];
    const float* gamma2 = (const float*)d_in[19];
    const float* beta2  = (const float*)d_in[20];
    const float* W1     = (const float*)d_in[21];
    const float* b1     = (const float*)d_in[22];
    const float* wdw    = (const float*)d_in[23];
    const float* bdw    = (const float*)d_in[24];
    const float* W2     = (const float*)d_in[25];
    const float* b2     = (const float*)d_in[26];
    float* out = (float*)d_out;

    // workspace layout (floats), with deliberate aliasing after lifetimes end
    float* wsf = (float*)d_ws;
    size_t o = 0;
    float* xnln  = wsf + o; o += SZ_;            // later reused as vbuf
    float* xn    = wsf + o; o += SZ_;            // later reused as ygate
    float* zb    = wsf + o; o += SZ_;            // later reused as out1
    float* ab    = wsf + o; o += SZ_;            // later reused as yn
    float* gb    = wsf + o; o += SZ_;
    float* kmean = wsf + o; o += (size_t)B_ * T_ * C_;
    float* ksm   = wsf + o; o += (size_t)B_ * T_ * C_;
    float* qm    = wsf + o; o += (size_t)B_ * C_;
    float* meanb = wsf + o; o += B_;
    float* rstdb = wsf + o; o += B_;
    float* logits= wsf + o; o += B_ * T_;
    float* wsoft = wsf + o; o += B_ * T_;
    float* h1    = wsf + o; o += HSZ_;
    float* h1d   = wsf + o; o += HSZ_;
    float* vbuf  = xnln;
    float* ygate = xn;
    float* out1  = zb;
    float* yn    = ab;

    const int EB = 256;
    const int gSZ = (SZ_ + EB - 1) / EB;
    const int gH  = (int)((HSZ_ + EB - 1) / EB);

    // 1) global-LN stats + q means
    stats_kernel<<<B_, 384, 0, stream>>>(x, qm, meanb, rstdb);
    // 2) LN1
    ln1_kernel<<<gSZ, EB, 0, stream>>>(x, meanb, rstdb, gamma1, beta1, xnln);
    // 3) positional depthwise conv
    posconv_kernel<<<gSZ, EB, 0, stream>>>(xnln, w_pos, b_pos, xn);
    // 4) z / a / g GEMMs (WMMA f16, 128x64 tiles)
    gemm_wmma_f16<0><<<dim3(C_ / 64, NROWS_ / 128), 256, 0, stream>>>(xn, W_in, b_in, nullptr, zb, NROWS_, C_, C_);
    gemm_wmma_f16<1><<<dim3(C_ / 64, NROWS_ / 128), 256, 0, stream>>>(xn, W_a, b_a, nullptr, ab, NROWS_, C_, C_);
    gemm_wmma_f16<2><<<dim3(C_ / 64, NROWS_ / 128), 256, 0, stream>>>(xn, W_g, b_g, nullptr, gb, NROWS_, C_, C_);
    // 5) per-t spatial means of hm_t * g   (h_t = (1-a^t) z closed form)
    tmeans_kernel<<<B_ * C_, 256, 0, stream>>>(ab, zb, gb, k_sp, b_sp, kmean);
    // 6) k = kmean @ W_out + b_out  (256-row WMMA GEMM)
    gemm_wmma_f16<0><<<dim3(C_ / 64, (B_ * T_) / 128), 256, 0, stream>>>(kmean, W_out, b_out, nullptr, ksm, B_ * T_, C_, C_);
    // 7) gate MLP + softmax over T
    gate_kernel<<<B_ * T_, GH_, 0, stream>>>(qm, ksm, Wg1, bg1, Wg2, bg2, logits);
    softmax_kernel<<<B_, 32, 0, stream>>>(logits, wsoft);
    // 8) v = z*(1 - sum w_t a^t); s = conv(v)+b_sp; ygate = s*g
    vmix_kernel<<<gSZ, EB, 0, stream>>>(ab, zb, wsoft, vbuf);
    comb_kernel<<<gSZ, EB, 0, stream>>>(vbuf, gb, k_sp, b_sp, ygate);
    // 9) out1 = x + ygate @ W_out + b_out
    gemm_wmma_f16<3><<<dim3(C_ / 64, NROWS_ / 128), 256, 0, stream>>>(ygate, W_out, b_out, x, out1, NROWS_, C_, C_);
    // 10) LN2
    ln2_kernel<<<NROWS_, 384, 0, stream>>>(out1, gamma2, beta2, yn);
    // 11) MLP: h1 = yn @ W1 + b1 ; h1d = gelu(dwconv(h1)+bdw) ; out = out1 + h1d @ W2 + b2
    gemm_wmma_f16<0><<<dim3(HID_ / 64, NROWS_ / 128), 256, 0, stream>>>(yn, W1, b1, nullptr, h1, NROWS_, C_, HID_);
    mlpconv_kernel<<<gH, EB, 0, stream>>>(h1, wdw, bdw, h1d);
    gemm_wmma_f16<3><<<dim3(C_ / 64, NROWS_ / 128), 256, 0, stream>>>(h1d, W2, b2, out1, out, NROWS_, HID_, C_);
}
